// RetrosynthesisSeq2SeqModel_7095285973378
// MI455X (gfx1250) — matrix operand order repaired and checked by reading, not compiled, not run on hardware
//
#include <hip/hip_runtime.h>

// ---------------------------------------------------------------------------
// Types for CDNA5 WMMA (wave32): bf16 A/B fragments (16 x bf16 = 8 dwords),
// f32 accumulator (8 floats).
// ---------------------------------------------------------------------------
typedef __attribute__((ext_vector_type(16))) __bf16 bf16x16;
typedef __attribute__((ext_vector_type(8)))  float  f32x8;

union Frag { bf16x16 v; unsigned int u[8]; };

__device__ __forceinline__ unsigned short f2bf(float x) {
  unsigned int u = __float_as_uint(x);
  u += 0x7FFFu + ((u >> 16) & 1u);   // round-to-nearest-even
  return (unsigned short)(u >> 16);
}
__device__ __forceinline__ float sigm(float x) { return 1.0f / (1.0f + __expf(-x)); }

// ---------------------------------------------------------------------------
// Weight prep: f32 [K][N] -> bf16 [N][K] (transposed so B-fragments are
// contiguous in K).
// ---------------------------------------------------------------------------
__global__ __launch_bounds__(256) void k_transpose_bf16(
    const float* __restrict__ in, unsigned short* __restrict__ out, int K, int N) {
  int i = blockIdx.x * blockDim.x + threadIdx.x;
  if (i >= K * N) return;
  int k = i / N, n = i - k * N;
  out[(size_t)n * K + k] = f2bf(in[i]);
}

__global__ __launch_bounds__(256) void k_convert_bf16(
    const float* __restrict__ in, unsigned short* __restrict__ out, int n) {
  int i = blockIdx.x * blockDim.x + threadIdx.x;
  if (i < n) out[i] = f2bf(in[i]);
}

// Embedding gather -> bf16 activations (EMB = 256)
__global__ __launch_bounds__(256) void k_embed(
    const int* __restrict__ tok, const float* __restrict__ emb,
    unsigned short* __restrict__ out, int ntok) {
  int i = blockIdx.x * blockDim.x + threadIdx.x;
  if (i >= ntok * 256) return;
  int t = i >> 8, j = i & 255;
  out[i] = f2bf(emb[(size_t)tok[t] * 256 + j]);
}

// ---------------------------------------------------------------------------
// Generic WMMA GEMM: C[M][N](f32) = A[M][K](bf16) * Bt[N][K](bf16)^T + bias.
// One wave per 16x16 C tile; K in chunks of 32 via v_wmma_f32_16x16x32_bf16.
// A-fragment layout (16-bit 16x32): lanes 0-15 rows, K base 0; lanes 16-31
// K base +8; VGPR v<4 -> K=kb+2v, v>=4 -> K=kb+16+2(v-4).
// B-fragment layout (16-bit 32x16): same K mapping as A (lanes 16-31 K base +8),
// packed 2 per dword.
// ---------------------------------------------------------------------------
__global__ __launch_bounds__(256) void k_wmma_gemm(
    const unsigned short* __restrict__ A,
    const unsigned short* __restrict__ Bt,
    const float* __restrict__ bias,
    float* __restrict__ C,
    int M, int N, int K) {
  int lane = threadIdx.x & 31;
  int wave = threadIdx.x >> 5;
  int tilesN = N >> 4;
  int tile = blockIdx.x * 8 + wave;
  if (tile >= (M >> 4) * tilesN) return;   // whole wave exits together (EXEC stays full)
  int tm = tile / tilesN, tn = tile - tm * tilesN;
  int l15 = lane & 15;
  int am = tm * 16 + l15;
  int bn = tn * 16 + l15;
  int kbA = (lane & 16) ? 8 : 0;
  int kbB = (lane & 16) ? 8 : 0;
  f32x8 acc = {0.f, 0.f, 0.f, 0.f, 0.f, 0.f, 0.f, 0.f};
  for (int kc = 0; kc < K; kc += 32) {
    Frag a, b;
    const unsigned short* ap = A + (size_t)am * K + kc + kbA;
    const unsigned short* bp = Bt + (size_t)bn * K + kc + kbB;
#pragma unroll
    for (int v = 0; v < 4; ++v) {
      a.u[v]     = *(const unsigned int*)(ap + 2 * v);
      a.u[v + 4] = *(const unsigned int*)(ap + 16 + 2 * v);
    }
#pragma unroll
    for (int v = 0; v < 4; ++v) { b.u[v] = *(const unsigned int*)(bp + 2 * v); b.u[v + 4] = *(const unsigned int*)(bp + 16 + 2 * v); }
    acc = __builtin_amdgcn_wmma_f32_16x16x32_bf16(false, a.v, false, b.v,
                                                  (short)0, acc, false, false);
  }
  int cn = tn * 16 + l15;
  int cm0 = tm * 16 + ((lane & 16) ? 8 : 0);
  float bv = bias ? bias[cn] : 0.f;
#pragma unroll
  for (int r = 0; r < 8; ++r) C[(size_t)(cm0 + r) * N + cn] = acc[r] + bv;
}

// ---------------------------------------------------------------------------
// Persistent single-workgroup LSTM over all T steps.
// 1024 threads = 32 waves; wave (mt,ut) owns rows mt*16..+15 x cols u0..u0+15
// of all FOUR gate tiles (N = g*256 + u0), so cell state c stays in registers.
// h (bf16) is shared through LDS (row stride 258 to avoid bank conflicts).
// xz = x@Wk + b precomputed in global.  2.  WMMAs/step/wave = 8 kchunks * 4.
// ---------------------------------------------------------------------------
__global__ __launch_bounds__(1024) void k_lstm_seq(
    const float* __restrict__ xz,            // [32][T][1024]
    const unsigned short* __restrict__ WrT,  // [1024][256] bf16
    const float* __restrict__ h0,            // [32][256] f32 or null
    const float* __restrict__ c0,            // [32][256] f32 or null
    float* __restrict__ hsF, int sF, int oF,            // f32 out (null ok)
    unsigned short* __restrict__ hsB, int sB, int oB,   // bf16 out (null ok)
    float* __restrict__ finH, float* __restrict__ finC, int sFin, int oFin,
    int T, int rev) {
  extern __shared__ unsigned short sh[];     // [32][258] bf16 h
  const int LDA = 258;
  int tid = threadIdx.x;
  int lane = tid & 31, wave = tid >> 5;
  int mt = wave >> 4;                // 0..1
  int u0 = (wave & 15) << 4;         // 0,16,...,240

  for (int i = tid; i < 32 * 256; i += 1024) {
    int b = i >> 8, u = i & 255;
    sh[b * LDA + u] = f2bf(h0 ? h0[i] : 0.f);
  }
  int l15 = lane & 15;
  int cn  = u0 + l15;                         // column within U
  int cm0 = mt * 16 + ((lane & 16) ? 8 : 0);  // C/D row base for this lane
  int am  = mt * 16 + l15;                    // A row for this lane
  int kbA = (lane & 16) ? 8 : 0;
  int kbB = (lane & 16) ? 8 : 0;

  float creg[8], hreg[8];
#pragma unroll
  for (int r = 0; r < 8; ++r) {
    creg[r] = c0 ? c0[(cm0 + r) * 256 + cn] : 0.f;
    hreg[r] = 0.f;
  }

  for (int t = 0; t < T; ++t) {
    int tt = rev ? (T - 1 - t) : t;
    __syncthreads();                           // h tile visible to all waves
    f32x8 acc[4];
#pragma unroll
    for (int g = 0; g < 4; ++g) acc[g] = (f32x8){0.f,0.f,0.f,0.f,0.f,0.f,0.f,0.f};

    for (int kc = 0; kc < 256; kc += 32) {
      Frag a;
      const unsigned short* ap = &sh[am * LDA + kc + kbA];
#pragma unroll
      for (int v = 0; v < 4; ++v) {
        a.u[v]     = *(const unsigned int*)(ap + 2 * v);
        a.u[v + 4] = *(const unsigned int*)(ap + 16 + 2 * v);
      }
#pragma unroll
      for (int g = 0; g < 4; ++g) {
        Frag b;
        int nn = (g << 8) + u0 + l15;
        const unsigned short* bp = WrT + (size_t)nn * 256 + kc + kbB;
#pragma unroll
        for (int v = 0; v < 4; ++v) { b.u[v] = *(const unsigned int*)(bp + 2 * v); b.u[v + 4] = *(const unsigned int*)(bp + 16 + 2 * v); }
        acc[g] = __builtin_amdgcn_wmma_f32_16x16x32_bf16(false, a.v, false, b.v,
                                                         (short)0, acc[g], false, false);
      }
    }
    // Gates (Keras order i,f,g,o) entirely in registers.
#pragma unroll
    for (int r = 0; r < 8; ++r) {
      int b = cm0 + r;
      const float* xp = xz + ((size_t)b * T + tt) * 1024;
      float iv = sigm(acc[0][r] + xp[cn]);
      float fv = sigm(acc[1][r] + xp[256 + cn]);
      float gv = tanhf(acc[2][r] + xp[512 + cn]);
      float ov = sigm(acc[3][r] + xp[768 + cn]);
      creg[r] = fv * creg[r] + iv * gv;
      float hv = ov * tanhf(creg[r]);
      hreg[r] = hv;
      if (hsF) hsF[((size_t)b * T + tt) * sF + oF + cn] = hv;
      if (hsB) hsB[((size_t)b * T + tt) * sB + oB + cn] = f2bf(hv);
    }
    __syncthreads();                           // all reads of old h done
#pragma unroll
    for (int r = 0; r < 8; ++r) sh[(cm0 + r) * LDA + cn] = f2bf(hreg[r]);
  }
  if (finH) {
#pragma unroll
    for (int r = 0; r < 8; ++r) finH[(cm0 + r) * sFin + oFin + cn] = hreg[r];
  }
  if (finC) {
#pragma unroll
    for (int r = 0; r < 8; ++r) finC[(cm0 + r) * sFin + oFin + cn] = creg[r];
  }
}

// ---------------------------------------------------------------------------
// Additive attention score: logits[b,td,te] = v . tanh(ep[b,te,:]+dp[b,td,:]) + vb
// One wave per output; lane-strided dot + shuffle reduce. (tanh inside the
// reduction forbids the matrix path.)
// ---------------------------------------------------------------------------
__global__ __launch_bounds__(256) void k_attn_score(
    const float* __restrict__ ep, const float* __restrict__ dp,
    const float* __restrict__ v, const float* __restrict__ vb,
    float* __restrict__ logits, int TD, int TE, int total) {
  int lane = threadIdx.x & 31;
  int idx = blockIdx.x * 8 + (threadIdx.x >> 5);
  if (idx >= total) return;
  int te = idx % TE;
  int td = (idx / TE) % TD;
  int b  = idx / (TE * TD);
  const float* epr = ep + ((size_t)b * TE + te) * 256;
  const float* dpr = dp + ((size_t)b * TD + td) * 256;
  float s = 0.f;
  for (int u = lane; u < 256; u += 32) s += tanhf(epr[u] + dpr[u]) * v[u];
#pragma unroll
  for (int o = 16; o > 0; o >>= 1) s += __shfl_xor(s, o, 32);
  if (lane == 0) logits[idx] = s + vb[0];
}

// In-place row softmax; one wave per row.
__global__ __launch_bounds__(256) void k_softmax_rows(float* __restrict__ x, int R, int C) {
  int lane = threadIdx.x & 31;
  int row = blockIdx.x * 8 + (threadIdx.x >> 5);
  if (row >= R) return;
  float* p = x + (size_t)row * C;
  float mx = -3.0e38f;
  for (int j = lane; j < C; j += 32) mx = fmaxf(mx, p[j]);
#pragma unroll
  for (int o = 16; o > 0; o >>= 1) mx = fmaxf(mx, __shfl_xor(mx, o, 32));
  float s = 0.f;
  for (int j = lane; j < C; j += 32) { float e = __expf(p[j] - mx); p[j] = e; s += e; }
#pragma unroll
  for (int o = 16; o > 0; o >>= 1) s += __shfl_xor(s, o, 32);
  float inv = 1.0f / s;
  for (int j = lane; j < C; j += 32) p[j] *= inv;
}

// context[b,td,j] = sum_te attn[b,td,te] * enc[b,te,j]   (j < 512)
__global__ __launch_bounds__(256) void k_context(
    const float* __restrict__ attn, const float* __restrict__ enc,
    float* __restrict__ ctx, int TD, int TE) {
  int i = blockIdx.x * blockDim.x + threadIdx.x;
  if (i >= 32 * TD * 512) return;
  int j  = i & 511;
  int td = (i >> 9) % TD;
  int b  = i / (512 * TD);
  const float* a = attn + ((size_t)b * TD + td) * TE;
  const float* e = enc + (size_t)b * TE * 512 + j;
  float s = 0.f;
  for (int te = 0; te < TE; ++te) s += a[te] * e[(size_t)te * 512];
  ctx[i] = s;
}

// concat [d | ctx] -> bf16 [rows][768]
__global__ __launch_bounds__(256) void k_concat_bf16(
    const float* __restrict__ d, const float* __restrict__ ctx,
    unsigned short* __restrict__ out, int rows) {
  int i = blockIdx.x * blockDim.x + threadIdx.x;
  if (i >= rows * 768) return;
  int r = i / 768, c = i - r * 768;
  float v = (c < 256) ? d[(size_t)r * 256 + c] : ctx[(size_t)r * 512 + (c - 256)];
  out[i] = f2bf(v);
}

// ---------------------------------------------------------------------------
// Host orchestration
// ---------------------------------------------------------------------------
extern "C" void kernel_launch(void* const* d_in, const int* in_sizes, int n_in,
                              void* d_out, int out_size, void* d_ws, size_t ws_size,
                              hipStream_t stream) {
  (void)in_sizes; (void)n_in; (void)out_size; (void)ws_size;
  const int Bc = 32, TEc = 100, TDc = 100;
  const int M = Bc * 100;      // 3200 rows for batch GEMMs

  // ---- input pointers (setup_inputs dict order, recursively) ----
  const int*   enc_tok = (const int*)d_in[0];
  const int*   dec_tok = (const int*)d_in[1];
  const float* emb_enc = (const float*)d_in[2];
  const float* emb_dec = (const float*)d_in[3];
  const float *Wk[8], *Wr[8], *bg[8];
  for (int i = 0; i < 8; ++i) {
    Wk[i] = (const float*)d_in[4 + 3 * i];
    Wr[i] = (const float*)d_in[5 + 3 * i];
    bg[i] = (const float*)d_in[6 + 3 * i];
  }
  const float* att1_W = (const float*)d_in[28];
  const float* att1_b = (const float*)d_in[29];
  const float* att2_W = (const float*)d_in[30];
  const float* att2_b = (const float*)d_in[31];
  const float* attv_W = (const float*)d_in[32];
  const float* attv_b = (const float*)d_in[33];
  const float* sh_W   = (const float*)d_in[34];
  const float* sh_b   = (const float*)d_in[35];
  const float* sc_W   = (const float*)d_in[36];
  const float* sc_b   = (const float*)d_in[37];
  const float* out_W  = (const float*)d_in[38];
  const float* out_b  = (const float*)d_in[39];

  // ---- bump allocator on workspace ----
  char* base = (char*)d_ws; size_t off = 0;
  auto alloc = [&](size_t bytes) -> void* {
    void* p = base + off; off = (off + bytes + 255) & ~(size_t)255; return p;
  };
  const int WkD[8] = {256, 256, 512, 512, 256, 256, 256, 256};
  unsigned short *WkT[8], *WrT[8];
  for (int i = 0; i < 8; ++i) WkT[i] = (unsigned short*)alloc((size_t)1024 * WkD[i] * 2);
  for (int i = 0; i < 8; ++i) WrT[i] = (unsigned short*)alloc((size_t)1024 * 256 * 2);
  unsigned short* att1T = (unsigned short*)alloc((size_t)256 * 512 * 2);
  unsigned short* att2T = (unsigned short*)alloc((size_t)256 * 256 * 2);
  unsigned short* shT   = (unsigned short*)alloc((size_t)256 * 512 * 2);
  unsigned short* scT   = (unsigned short*)alloc((size_t)256 * 512 * 2);
  unsigned short* outT  = (unsigned short*)alloc((size_t)128 * 768 * 2);
  unsigned short* xenc  = (unsigned short*)alloc((size_t)M * 256 * 2);
  unsigned short* xdec  = (unsigned short*)alloc((size_t)M * 256 * 2);
  float*          xz    = (float*)alloc((size_t)M * 1024 * 4);
  unsigned short* e1cat = (unsigned short*)alloc((size_t)M * 512 * 2);
  float*          e2catF= (float*)alloc((size_t)M * 512 * 4);
  unsigned short* e2catB= (unsigned short*)alloc((size_t)M * 512 * 2);
  float* stateH = (float*)alloc((size_t)Bc * 512 * 4);
  float* stateC = (float*)alloc((size_t)Bc * 512 * 4);
  unsigned short* stateHb = (unsigned short*)alloc((size_t)Bc * 512 * 2);
  unsigned short* stateCb = (unsigned short*)alloc((size_t)Bc * 512 * 2);
  float* h0 = (float*)alloc((size_t)Bc * 256 * 4);
  float* c0 = (float*)alloc((size_t)Bc * 256 * 4);
  float*          decAF = (float*)alloc((size_t)M * 256 * 4);
  unsigned short* decAB = (unsigned short*)alloc((size_t)M * 256 * 2);
  float*          decBF = (float*)alloc((size_t)M * 256 * 4);
  unsigned short* decBB = (unsigned short*)alloc((size_t)M * 256 * 2);
  float* encP   = (float*)alloc((size_t)M * 256 * 4);
  float* decP   = (float*)alloc((size_t)M * 256 * 4);
  float* logits = (float*)alloc((size_t)Bc * TDc * TEc * 4);
  float* ctx    = (float*)alloc((size_t)M * 512 * 4);
  unsigned short* cat = (unsigned short*)alloc((size_t)M * 768 * 2);

  // ---- launch helpers ----
  auto tr = [&](const float* s, unsigned short* d, int K, int N) {
    int n = K * N;
    k_transpose_bf16<<<(n + 255) / 256, 256, 0, stream>>>(s, d, K, N);
  };
  auto gemm = [&](const unsigned short* A, const unsigned short* Bt, const float* bias,
                  float* C, int m, int n, int k) {
    int tiles = (m / 16) * (n / 16);
    k_wmma_gemm<<<(tiles + 7) / 8, 256, 0, stream>>>(A, Bt, bias, C, m, n, k);
  };
  const size_t lstmLds = (size_t)32 * 258 * 2;
  auto lstm = [&](const float* xzp, const unsigned short* WrTp,
                  const float* h0p, const float* c0p,
                  float* hsF, int sF, int oF,
                  unsigned short* hsB, int sB, int oB,
                  float* fH, float* fC, int sFin, int oFin, int T, int rev) {
    k_lstm_seq<<<1, 1024, lstmLds, stream>>>(xzp, WrTp, h0p, c0p,
                                             hsF, sF, oF, hsB, sB, oB,
                                             fH, fC, sFin, oFin, T, rev);
  };

  // ---- weight prep ----
  for (int i = 0; i < 8; ++i) { tr(Wk[i], WkT[i], WkD[i], 1024); tr(Wr[i], WrT[i], 256, 1024); }
  tr(att1_W, att1T, 512, 256);
  tr(att2_W, att2T, 256, 256);
  tr(sh_W, shT, 512, 256);
  tr(sc_W, scT, 512, 256);
  tr(out_W, outT, 768, 128);

  // ---- embeddings ----
  k_embed<<<(M * 256 + 255) / 256, 256, 0, stream>>>(enc_tok, emb_enc, xenc, M);
  k_embed<<<(M * 256 + 255) / 256, 256, 0, stream>>>(dec_tok, emb_dec, xdec, M);

  // ---- encoder layer 1 (bi) ----
  gemm(xenc, WkT[0], bg[0], xz, M, 1024, 256);
  lstm(xz, WrT[0], nullptr, nullptr, nullptr, 0, 0, e1cat, 512, 0,
       nullptr, nullptr, 0, 0, TEc, 0);
  gemm(xenc, WkT[1], bg[1], xz, M, 1024, 256);
  lstm(xz, WrT[1], nullptr, nullptr, nullptr, 0, 0, e1cat, 512, 256,
       nullptr, nullptr, 0, 0, TEc, 1);

  // ---- encoder layer 2 (bi) + final states ----
  gemm(e1cat, WkT[2], bg[2], xz, M, 1024, 512);
  lstm(xz, WrT[2], nullptr, nullptr, e2catF, 512, 0, e2catB, 512, 0,
       stateH, stateC, 512, 0, TEc, 0);
  gemm(e1cat, WkT[3], bg[3], xz, M, 1024, 512);
  lstm(xz, WrT[3], nullptr, nullptr, e2catF, 512, 256, e2catB, 512, 256,
       stateH, stateC, 512, 256, TEc, 1);

  // ---- initial decoder states ----
  k_convert_bf16<<<(Bc * 512 + 255) / 256, 256, 0, stream>>>(stateH, stateHb, Bc * 512);
  k_convert_bf16<<<(Bc * 512 + 255) / 256, 256, 0, stream>>>(stateC, stateCb, Bc * 512);
  gemm(stateHb, shT, sh_b, h0, Bc, 256, 512);
  gemm(stateCb, scT, sc_b, c0, Bc, 256, 512);

  // ---- decoder stack ----
  gemm(xdec, WkT[4], bg[4], xz, M, 1024, 256);
  lstm(xz, WrT[4], h0, c0, decAF, 256, 0, decAB, 256, 0, nullptr, nullptr, 0, 0, TDc, 0);
  gemm(decAB, WkT[5], bg[5], xz, M, 1024, 256);
  lstm(xz, WrT[5], nullptr, nullptr, decBF, 256, 0, decBB, 256, 0, nullptr, nullptr, 0, 0, TDc, 0);
  gemm(decBB, WkT[6], bg[6], xz, M, 1024, 256);
  lstm(xz, WrT[6], nullptr, nullptr, decAF, 256, 0, decAB, 256, 0, nullptr, nullptr, 0, 0, TDc, 0);
  gemm(decAB, WkT[7], bg[7], xz, M, 1024, 256);
  lstm(xz, WrT[7], nullptr, nullptr, decBF, 256, 0, decBB, 256, 0, nullptr, nullptr, 0, 0, TDc, 0);
  // d = decB (f32 decBF, bf16 decBB)

  // ---- attention ----
  gemm(e2catB, att1T, att1_b, encP, M, 256, 512);
  gemm(decBB, att2T, att2_b, decP, M, 256, 256);
  int total = Bc * TDc * TEc;
  k_attn_score<<<(total + 7) / 8, 256, 0, stream>>>(encP, decP, attv_W, attv_b,
                                                    logits, TDc, TEc, total);
  k_softmax_rows<<<(Bc * TDc + 7) / 8, 256, 0, stream>>>(logits, Bc * TDc, TEc);
  int nctx = Bc * TDc * 512;
  k_context<<<(nctx + 255) / 256, 256, 0, stream>>>(logits, e2catF, ctx, TDc, TEc);

  // ---- output projection + softmax ----
  int ncat = M * 768;
  k_concat_bf16<<<(ncat + 255) / 256, 256, 0, stream>>>(decBF, ctx, cat, M);
  gemm(cat, outT, out_b, (float*)d_out, M, 128, 768);
  k_softmax_rows<<<(M + 7) / 8, 256, 0, stream>>>((float*)d_out, M, 128);
}
